// ValueNetwork_25589415150019
// MI455X (gfx1250) — matrix operand order repaired and checked by reading, not compiled, not run on hardware
//
#include <hip/hip_runtime.h>
#include <hip/hip_bf16.h>
#include <math.h>

typedef __attribute__((ext_vector_type(16))) _Float16 v16h;
typedef __attribute__((ext_vector_type(8)))  _Float16 v8h;
typedef __attribute__((ext_vector_type(8)))  float    v8f;

#define BATCH 4096
#define NROW  64
#define FDIM  13
#define DDIM  128
#define KPAD  32
#define SELFD 6

// f16 weight workspace layout (units = halves)
#define OFF_PHI2 0
#define OFF_PHI3 4096
#define OFF_Q    8192
#define OFF_K    24576
#define OFF_V    40960
#define WS_HALVES 57344

__device__ __forceinline__ v16h cat8(v8h lo, v8h hi) {
  return __builtin_shufflevector(lo, hi, 0,1,2,3,4,5,6,7,8,9,10,11,12,13,14,15);
}

// ---------------------------------------------------------------------------
// One-time fp32 -> fp16 weight conversion into d_ws (row-major (out,in),
// phi2/phi3 zero-padded from K=13 to K=32 so the GEMM stays exact).
// ---------------------------------------------------------------------------
__global__ void convert_weights_kernel(const float* __restrict__ phi2_w,
                                       const float* __restrict__ phi3_w,
                                       const float* __restrict__ q_w,
                                       const float* __restrict__ k_w,
                                       const float* __restrict__ v_w,
                                       _Float16* __restrict__ ws) {
  int idx = blockIdx.x * 256 + threadIdx.x;
  if (idx >= WS_HALVES) return;
  if (idx < 4096) {
    int r = idx >> 5, c = idx & 31;
    ws[OFF_PHI2 + idx] = (c < FDIM) ? (_Float16)phi2_w[r * FDIM + c] : (_Float16)0.0f;
  } else if (idx < 8192) {
    int i = idx - 4096; int r = i >> 5, c = i & 31;
    ws[OFF_PHI3 + i] = (c < FDIM) ? (_Float16)phi3_w[r * FDIM + c] : (_Float16)0.0f;
  } else if (idx < 24576) {
    ws[idx] = (_Float16)q_w[idx - 8192];
  } else if (idx < 40960) {
    ws[idx] = (_Float16)k_w[idx - 24576];
  } else {
    ws[idx] = (_Float16)v_w[idx - 40960];
  }
}

// ---------------------------------------------------------------------------
// Fully fused value network: one workgroup (8 wave32) per batch element.
// WMMA convention here: A = weight tile (n x K), B = activation^T (K x m),
// D[n][m] -> each lane holds 8 n-contiguous results => packed b128 LDS stores.
// ---------------------------------------------------------------------------
__global__ void __launch_bounds__(256)
value_net_kernel(const float* __restrict__ state,
                 const float* __restrict__ phi1_w, const float* __restrict__ phi1_b,
                 const float* __restrict__ phi2_b, const float* __restrict__ phi3_b,
                 const float* __restrict__ q_b,    const float* __restrict__ k_b,
                 const float* __restrict__ v_b,
                 const float* __restrict__ final_w, const float* __restrict__ final_b,
                 const _Float16* __restrict__ wsh,
                 float* __restrict__ out) {
  __shared__ _Float16 sA [NROW * KPAD];   // state, f16, K padded (row-major)   4 KB
  __shared__ _Float16 sH2[NROW * DDIM];   // relu(state@phi2ᵀ+b)  [m][n]      16 KB
  __shared__ _Float16 sH3[NROW * DDIM];   // relu(state@phi3ᵀ+b)  [m][n]      16 KB
  __shared__ _Float16 sQ [NROW * DDIM];   // attn_q (relu'd)      [m][n]      16 KB
  __shared__ _Float16 sK [NROW * DDIM];   // attn_k (relu'd)      [m][n]      16 KB
  __shared__ _Float16 sV [NROW * DDIM];   // attn_v               [m][n]      16 KB
  __shared__ float qsum[DDIM];
  __shared__ float eh[DDIM];
  __shared__ float part[256];
  __shared__ float scoresL[NROW];
  __shared__ float wexp[NROW];
  __shared__ float wnorm[NROW];

  const int b    = blockIdx.x;
  const int tid  = threadIdx.x;
  const int lane = tid & 31;
  const int wave = tid >> 5;
  const int half = lane >> 4;   // 16-lane half of the wave
  const int r    = lane & 15;

  // ---- Phase 0: stage state into LDS (f16, zero-padded K) + e_h -----------
  for (int i = tid; i < NROW * KPAD; i += 256) sA[i] = (_Float16)0.0f;
  __syncthreads();
  const float* stb = state + (size_t)b * NROW * FDIM;
  for (int i = tid; i < NROW * FDIM; i += 256) {
    int row = i / FDIM, col = i - row * FDIM;
    sA[row * KPAD + col] = (_Float16)stb[i];
  }
  if (tid < DDIM) {            // e_h = phi1_w @ state[b,0,:6] + phi1_b
    float s = phi1_b[tid];
#pragma unroll
    for (int j = 0; j < SELFD; ++j) s += stb[j] * phi1_w[tid * SELFD + j];
    eh[tid] = s;
  }
  __syncthreads();

  // ---- Phase 1: h2/h3 = relu(state @ phiᵀ + b). 16 jobs = {h2,h3} x 8 n-strips.
#pragma unroll
  for (int jj = 0; jj < 2; ++jj) {
    const int job = wave + jj * 8;
    const int mat = job >> 3;                 // 0 = h2, 1 = h3
    const int n0  = (job & 7) * 16;
    const _Float16* W    = wsh + (mat ? OFF_PHI3 : OFF_PHI2);
    const float*    bias = mat ? phi3_b : phi2_b;
    _Float16*       dst  = mat ? sH3 : sH2;
    // A fragment = weight tile rows n0..n0+15, K=0..31 (lane r -> row n0+r)
    const _Float16* ap = W + (n0 + r) * KPAD + half * 8;
    const v16h af = cat8(*(const v8h*)ap, *(const v8h*)(ap + 16));
    const v8f  bv = *(const v8f*)(bias + n0 + 8 * half);   // per-vgpr bias
#pragma unroll
    for (int mtp = 0; mtp < 4; mtp += 2) {    // M-tile pairs: 2 indep acc chains
      v8f acc[2];
      int m0s[2];
#pragma unroll
      for (int u = 0; u < 2; ++u) {
        const int m0 = (mtp + u) * 16; m0s[u] = m0;
        const _Float16* bp = sA + (m0 + r) * KPAD + half * 8;
        const v16h bf = cat8(*(const v8h*)bp, *(const v8h*)(bp + 16));
        v8f z = {};
        acc[u] = __builtin_amdgcn_wmma_f32_16x16x32_f16(false, af, false, bf,
                                                        (short)0, z, false, false);
      }
#pragma unroll
      for (int u = 0; u < 2; ++u) {
        v8h pk;
#pragma unroll
        for (int v = 0; v < 8; ++v) {
          float x = acc[u][v] + bv[v];
          pk[v] = (_Float16)(x > 0.0f ? x : 0.0f);
        }
        *(v8h*)(dst + (m0s[u] + r) * DDIM + n0 + 8 * half) = pk;  // packed 16B
      }
    }
  }
  __syncthreads();

  // ---- Phase 2: Q/K/V. 24 jobs = {Q,K,V} x 8 n-strips, K=128 in 4 steps ---
#pragma unroll
  for (int jj = 0; jj < 3; ++jj) {
    const int job = wave + jj * 8;
    const int mat = job >> 3;                 // 0=Q 1=K 2=V
    const int n0  = (job & 7) * 16;
    const _Float16* W    = wsh + (mat == 0 ? OFF_Q : (mat == 1 ? OFF_K : OFF_V));
    const float*    bias = (mat == 0 ? q_b : (mat == 1 ? k_b : v_b));
    const _Float16* srcH = (mat == 2) ? sH2 : sH3;
    _Float16*       dst  = (mat == 0 ? sQ : (mat == 1 ? sK : sV));
    v16h af[4];                               // hoist weight A-frags over M-tiles
#pragma unroll
    for (int kt = 0; kt < 4; ++kt) {
      const _Float16* ap = W + (n0 + r) * DDIM + kt * 32 + half * 8;
      af[kt] = cat8(*(const v8h*)ap, *(const v8h*)(ap + 16));
    }
    const v8f bv = *(const v8f*)(bias + n0 + 8 * half);
#pragma unroll
    for (int mtp = 0; mtp < 4; mtp += 2) {    // pairs: interleave 2 acc chains
      const int m0a = mtp * 16, m0b = (mtp + 1) * 16;
      v8f acc0 = {}, acc1 = {};
#pragma unroll
      for (int kt = 0; kt < 4; ++kt) {
        const _Float16* bp0 = srcH + (m0a + r) * DDIM + kt * 32 + half * 8;
        const v16h bf0 = cat8(*(const v8h*)bp0, *(const v8h*)(bp0 + 16));
        const _Float16* bp1 = srcH + (m0b + r) * DDIM + kt * 32 + half * 8;
        const v16h bf1 = cat8(*(const v8h*)bp1, *(const v8h*)(bp1 + 16));
        acc0 = __builtin_amdgcn_wmma_f32_16x16x32_f16(false, af[kt], false, bf0,
                                                      (short)0, acc0, false, false);
        acc1 = __builtin_amdgcn_wmma_f32_16x16x32_f16(false, af[kt], false, bf1,
                                                      (short)0, acc1, false, false);
      }
      const bool do_relu = (mat != 2);        // attn_v has no relu
      v8h pk0, pk1;
#pragma unroll
      for (int v = 0; v < 8; ++v) {
        float x0 = acc0[v] + bv[v];
        float x1 = acc1[v] + bv[v];
        if (do_relu) { x0 = x0 > 0.0f ? x0 : 0.0f; x1 = x1 > 0.0f ? x1 : 0.0f; }
        pk0[v] = (_Float16)x0;
        pk1[v] = (_Float16)x1;
      }
      *(v8h*)(dst + (m0a + r) * DDIM + n0 + 8 * half) = pk0;
      *(v8h*)(dst + (m0b + r) * DDIM + n0 + 8 * half) = pk1;
    }
  }
  __syncthreads();

  // ---- Phase 3: qsum[n] = sum_m attn_q[m][n] ------------------------------
  {
    const int n = tid & 127, j = tid >> 7;
    float s = 0.0f;
    for (int m = j * 32; m < j * 32 + 32; ++m) s += (float)sQ[m * DDIM + n];
    part[tid] = s;
  }
  __syncthreads();
  if (tid < DDIM) qsum[tid] = part[tid] + part[tid + 128];
  __syncthreads();

  // ---- Phase 4: scores[m] = relu(temp * qsum . k_m) (k_m is contiguous) ---
  {
    const int m = tid & 63, q4 = tid >> 6;
    float s = 0.0f;
    for (int n = q4 * 32; n < q4 * 32 + 32; ++n)
      s += qsum[n] * (float)sK[m * DDIM + n];
    part[tid] = s;
  }
  __syncthreads();
  if (tid < 64) {
    float s = part[tid] + part[tid + 64] + part[tid + 128] + part[tid + 192];
    s *= 0.08838834764831845f;                // 1/sqrt(128)
    scoresL[tid] = s > 0.0f ? s : 0.0f;
  }
  __syncthreads();

  // ---- Phase 5: softmax over 64 -------------------------------------------
  if (tid < 64) {
    float mx = -1e30f;
    for (int i = 0; i < 64; ++i) mx = fmaxf(mx, scoresL[i]);
    wexp[tid] = __expf(scoresL[tid] - mx);
  }
  __syncthreads();
  if (tid < 64) {
    float ssum = 0.0f;
    for (int i = 0; i < 64; ++i) ssum += wexp[i];
    wnorm[tid] = wexp[tid] / ssum;
  }
  __syncthreads();

  // ---- Phase 6: e_g = Vᵀ @ w, final dot, block reduce ---------------------
  float partial = 0.0f;
  if (tid < DDIM) {
    float g = 0.0f;
    for (int m = 0; m < NROW; ++m) g += wnorm[m] * (float)sV[m * DDIM + tid];
    partial = g * final_w[tid] + eh[tid] * final_w[DDIM + tid];
  }
  part[tid] = partial;
  __syncthreads();
  for (int s = 128; s >= 1; s >>= 1) {
    if (tid < s) part[tid] += part[tid + s];
    __syncthreads();
  }
  if (tid == 0) out[b] = part[0] + final_b[0];
}

// ---------------------------------------------------------------------------
extern "C" void kernel_launch(void* const* d_in, const int* in_sizes, int n_in,
                              void* d_out, int out_size, void* d_ws, size_t ws_size,
                              hipStream_t stream) {
  (void)in_sizes; (void)n_in; (void)out_size; (void)ws_size;
  const float* state   = (const float*)d_in[0];
  const float* phi1_w  = (const float*)d_in[1];
  const float* phi1_b  = (const float*)d_in[2];
  const float* phi2_w  = (const float*)d_in[3];
  const float* phi2_b  = (const float*)d_in[4];
  const float* phi3_w  = (const float*)d_in[5];
  const float* phi3_b  = (const float*)d_in[6];
  const float* q_w     = (const float*)d_in[7];
  const float* q_b     = (const float*)d_in[8];
  const float* k_w     = (const float*)d_in[9];
  const float* k_b     = (const float*)d_in[10];
  const float* v_w     = (const float*)d_in[11];
  const float* v_b     = (const float*)d_in[12];
  const float* final_w = (const float*)d_in[13];
  const float* final_b = (const float*)d_in[14];
  _Float16* wsh = (_Float16*)d_ws;
  float* out = (float*)d_out;

  convert_weights_kernel<<<(WS_HALVES + 255) / 256, 256, 0, stream>>>(
      phi2_w, phi3_w, q_w, k_w, v_w, wsh);
  value_net_kernel<<<BATCH, 256, 0, stream>>>(
      state, phi1_w, phi1_b, phi2_b, phi3_b, q_b, k_b, v_b,
      final_w, final_b, wsh, out);
}
